// LoRAAttention_84121229460045
// MI455X (gfx1250) — matrix operand order repaired and checked by reading, not compiled, not run on hardware
//
#include <hip/hip_runtime.h>
#include <cmath>

typedef __attribute__((ext_vector_type(16))) _Float16 v16h;
typedef __attribute__((ext_vector_type(8)))  _Float16 v8h;
typedef __attribute__((ext_vector_type(4)))  _Float16 v4h;
typedef __attribute__((ext_vector_type(8)))  float    v8f;

#define DM   1024
#define SEQ  2048
#define NB   2
#define NH   16
#define DKH  64
#define MTOK 4096   // NB*SEQ
#define QW   (SEQ / 32)   // mask words per key row

__device__ __forceinline__ v16h cat8(v8h a, v8h b) {
  return __builtin_shufflevector(a, b, 0,1,2,3,4,5,6,7,8,9,10,11,12,13,14,15);
}

__device__ __forceinline__ v8f wmma32(v16h a, v16h b, v8f c) {
  // D = A(16x32,f16) * B(32x16,f16) + C(16x16,f32)
  return __builtin_amdgcn_wmma_f32_16x16x32_f16(false, a, false, b, (short)0, c,
                                                false, false);
}

// A-fragment 16x32 (f16): row m = lane%16; lanes 0-15 hold K 0..7,16..23,
// lanes 16-31 hold K 8..15,24..31 (ISA 7.12.2). Source row-major, leading dim ld.
__device__ __forceinline__ v16h load_frag_a(const _Float16* __restrict__ base,
                                            int ld, int k0) {
  int lane = threadIdx.x & 31;
  int m = lane & 15, hi = lane >> 4;
  const _Float16* p = base + (size_t)m * ld + k0 + 8 * hi;
  v8h lo = *(const v8h*)(p);
  v8h hh = *(const v8h*)(p + 16);
  return cat8(lo, hh);
}

// B-fragment 32x16 (f16): B[k][n] = src[n*ld + k] (n-th row of row-major NxK src).
// Lane holds col n = lane%16; lanes 0-15 hold K 0..7,16..23, lanes 16-31 hold K 8..15,24..31.
__device__ __forceinline__ v16h load_frag_b(const _Float16* __restrict__ base,
                                            int ld, int k0) {
  int lane = threadIdx.x & 31;
  int n = lane & 15, hi = lane >> 4;
  const _Float16* p = base + (size_t)n * ld + k0 + 8 * hi;
  v8h lo = *(const v8h*)(p);
  v8h hh = *(const v8h*)(p + 16);
  return cat8(lo, hh);
}

// ---------------- W' = W + 2 * B @ A  (fold LoRA), f32 -> f16 ----------------
__global__ void fuse_w(const float* __restrict__ W, const float* __restrict__ A,
                       const float* __restrict__ Bm, _Float16* __restrict__ out) {
  int idx = blockIdx.x * 256 + threadIdx.x;   // over 1024*1024
  int n = idx >> 10, k = idx & 1023;
  float acc = W[idx];
#pragma unroll
  for (int r = 0; r < 8; ++r) acc += 2.0f * Bm[n * 8 + r] * A[r * 1024 + k];
  out[idx] = (_Float16)acc;
}

// ---------------- f32 -> f16 convert (4 elems/thread) ----------------
__global__ void cvt_f16(const float* __restrict__ in, _Float16* __restrict__ out) {
  int i = (blockIdx.x * 256 + threadIdx.x) * 4;
  float4 v = *(const float4*)(in + i);
  v4h o = { (_Float16)v.x, (_Float16)v.y, (_Float16)v.z, (_Float16)v.w };
  *(v4h*)(out + i) = o;
}

// ------- pack mask to transposed bitmask: mtb[(b*SEQ+k)*QW + q/32] bit(q%32) --
__global__ void pack_mask(const int* __restrict__ mask, unsigned* __restrict__ mtb) {
  int k  = blockIdx.x * 256 + threadIdx.x;   // 0..SEQ-1
  int qw = blockIdx.y;                        // 0..QW-1
  int b  = blockIdx.z;
  const int* mp = mask + ((size_t)b * SEQ + (size_t)qw * 32) * SEQ + k;
  unsigned w = 0;
#pragma unroll
  for (int j = 0; j < 32; ++j)
    w |= (mp[(size_t)j * SEQ] != 0 ? 1u : 0u) << j;
  mtb[((size_t)b * SEQ + k) * QW + qw] = w;
}

// ---------------- NT GEMM: C[MxN] = A[MxK] * B[NxK]^T ----------------
// 8 waves (4 along M, 2 along N); wave computes 32x64; block tile 128x128.
template <int F32OUT>
__global__ void gemm_nt(const _Float16* __restrict__ A, const _Float16* __restrict__ Bm,
                        float* __restrict__ Cf, _Float16* __restrict__ Ch,
                        int M, int N, int K) {
  int w = threadIdx.x >> 5;
  int wm = w >> 1, wn = w & 1;
  int m0 = blockIdx.y * 128 + wm * 32;
  int n0 = blockIdx.x * 128 + wn * 64;
  v8f acc[2][4];
#pragma unroll
  for (int i = 0; i < 2; ++i)
#pragma unroll
    for (int j = 0; j < 4; ++j) acc[i][j] = {};
  const _Float16* Ar[2] = { A + (size_t)m0 * K, A + (size_t)(m0 + 16) * K };
  const _Float16* Br[4] = { Bm + (size_t)n0 * K,        Bm + (size_t)(n0 + 16) * K,
                            Bm + (size_t)(n0 + 32) * K, Bm + (size_t)(n0 + 48) * K };
  int lane = threadIdx.x & 31, ln = lane & 15, hi = lane >> 4;
  for (int k0 = 0; k0 < K; k0 += 32) {
    if (k0 + 32 < K) {  // prefetch next k-slice (global_prefetch_b8)
      __builtin_prefetch(Ar[0] + (size_t)ln * K + k0 + 32 + 8 * hi, 0, 1);
      __builtin_prefetch(Br[0] + (size_t)ln * K + k0 + 32 + 16 * hi, 0, 1);
      __builtin_prefetch(Br[2] + (size_t)ln * K + k0 + 32 + 16 * hi, 0, 1);
    }
    v16h a0 = load_frag_a(Ar[0], K, k0);
    v16h a1 = load_frag_a(Ar[1], K, k0);
#pragma unroll
    for (int j = 0; j < 4; ++j) {
      v16h bj = load_frag_b(Br[j], K, k0);
      acc[0][j] = wmma32(a0, bj, acc[0][j]);
      acc[1][j] = wmma32(a1, bj, acc[1][j]);
    }
  }
#pragma unroll
  for (int i = 0; i < 2; ++i)
#pragma unroll
    for (int j = 0; j < 4; ++j) {
#pragma unroll
      for (int v = 0; v < 8; ++v) {
        int row = m0 + 16 * i + v + 8 * hi;
        int col = n0 + 16 * j + ln;
        if constexpr (F32OUT) Cf[(size_t)row * N + col] = acc[i][j][v];
        else                  Ch[(size_t)row * N + col] = (_Float16)acc[i][j][v];
      }
    }
}

// ------- LDS-tiled V transpose: VT[b*DM+d][s] = V[b*SEQ+s][d], 64x64 tiles ----
__global__ void transpose_v(const _Float16* __restrict__ V, _Float16* __restrict__ VT) {
  __shared__ __align__(16) _Float16 t[64][72];   // pad: 144B rows, 8B-aligned cols
  int b = blockIdx.z;
  int s0 = blockIdx.x * 64, d0 = blockIdx.y * 64;
  int c4 = (threadIdx.x & 15) * 4;   // 0..60 step 4
  int r0 = threadIdx.x >> 4;         // 0..15
#pragma unroll
  for (int rr = 0; rr < 64; rr += 16) {
    v4h x = *(const v4h*)&V[((size_t)b * SEQ + s0 + r0 + rr) * DM + d0 + c4];
    *(v4h*)&t[r0 + rr][c4] = x;
  }
  __syncthreads();
#pragma unroll
  for (int rr = 0; rr < 64; rr += 16) {
    int d = r0 + rr;
    v4h y = { t[c4 + 0][d], t[c4 + 1][d], t[c4 + 2][d], t[c4 + 3][d] };
    *(v4h*)&VT[((size_t)b * DM + d0 + d) * SEQ + s0 + c4] = y;
  }
}

// ---------------- Flash attention: one wave = 16 query rows of one (b,h) ----
// (16 rows/wave stays within the register budget -> no scratch spills)
__global__ void attn(const _Float16* __restrict__ Q, const _Float16* __restrict__ Km,
                     const _Float16* __restrict__ VT, const unsigned* __restrict__ mtb,
                     _Float16* __restrict__ ctx) {
  __shared__ __align__(16) _Float16 plds[8][16 * 32];  // 1KB per wave
  int wave = threadIdx.x >> 5;
  int lane = threadIdx.x & 31;
  int ln = lane & 15, hi = lane >> 4;
  int b = blockIdx.z, h = blockIdx.y;
  int q0 = blockIdx.x * 128 + wave * 16;
  int qsh = (q0 & 16) + 8 * hi;   // bit offset of this lane's first row in mask word

  // Q fragment, pre-scaled by 1/sqrt(d_k) = 1/8
  const _Float16* qbase = Q + ((size_t)b * SEQ + q0) * DM + h * DKH;
  v16h qf0 = load_frag_a(qbase, DM, 0);
  v16h qf1 = load_frag_a(qbase, DM, 32);
  const _Float16 qs = (_Float16)0.125f;
#pragma unroll
  for (int i = 0; i < 16; ++i) { qf0[i] *= qs; qf1[i] *= qs; }

  v8f acc0 = {}, acc1 = {}, acc2 = {}, acc3 = {};
  float mrun[8], lrun[8];
#pragma unroll
  for (int v = 0; v < 8; ++v) { mrun[v] = -INFINITY; lrun[v] = 0.f; }

  const _Float16* vtb = VT + ((size_t)b * DM + h * DKH) * SEQ;
  const unsigned* mrow = mtb + (size_t)b * SEQ * QW + (q0 >> 5);

  for (int kb = 0; kb < SEQ; kb += 32) {
    const _Float16* kb0 = Km + ((size_t)b * SEQ + kb) * DM + h * DKH;
    const _Float16* kb1 = kb0 + (size_t)16 * DM;
    v8f s0 = {}, s1 = {};
    s0 = wmma32(qf0, load_frag_b(kb0, DM, 0),  s0);
    s0 = wmma32(qf1, load_frag_b(kb0, DM, 32), s0);
    s1 = wmma32(qf0, load_frag_b(kb1, DM, 0),  s1);
    s1 = wmma32(qf1, load_frag_b(kb1, DM, 32), s1);

    // mask via packed transposed bitmask (1 word per key-tile, bit = q row)
    unsigned w0 = mrow[((size_t)kb + ln) * QW];
    unsigned w1 = mrow[((size_t)kb + 16 + ln) * QW];
#pragma unroll
    for (int v = 0; v < 8; ++v) {
      if (!((w0 >> (qsh + v)) & 1u)) s0[v] = -1e9f;
      if (!((w1 >> (qsh + v)) & 1u)) s1[v] = -1e9f;
    }

    // streaming softmax
#pragma unroll
    for (int v = 0; v < 8; ++v) {
      float mx = fmaxf(s0[v], s1[v]);
      mx = fmaxf(mx, __shfl_xor(mx, 1, 16));
      mx = fmaxf(mx, __shfl_xor(mx, 2, 16));
      mx = fmaxf(mx, __shfl_xor(mx, 4, 16));
      mx = fmaxf(mx, __shfl_xor(mx, 8, 16));
      float mnew = fmaxf(mrun[v], mx);
      float alpha = __expf(mrun[v] - mnew);
      s0[v] = __expf(s0[v] - mnew);
      s1[v] = __expf(s1[v] - mnew);
      float ps = s0[v] + s1[v];
      ps += __shfl_xor(ps, 1, 16);
      ps += __shfl_xor(ps, 2, 16);
      ps += __shfl_xor(ps, 4, 16);
      ps += __shfl_xor(ps, 8, 16);
      lrun[v] = lrun[v] * alpha + ps;
      mrun[v] = mnew;
      acc0[v] *= alpha; acc1[v] *= alpha;
      acc2[v] *= alpha; acc3[v] *= alpha;
    }

    // C-layout -> A-fragment relayout of P through per-wave LDS
    _Float16* pw = &plds[wave][0];
#pragma unroll
    for (int v = 0; v < 8; ++v) {
      int row = v + 8 * hi;
      pw[row * 32 + ln]      = (_Float16)s0[v];
      pw[row * 32 + 16 + ln] = (_Float16)s1[v];
    }
    v16h pf;
    {
      const _Float16* pr = pw + ln * 32 + 8 * hi;
      v8h lo = *(const v8h*)(pr);
      v8h hh = *(const v8h*)(pr + 16);
      pf = cat8(lo, hh);
    }

    // ctx += P(16x32) @ V(32x64)  (Vt rows are d, contiguous in s)
    acc0 = wmma32(pf, load_frag_b(vtb,            SEQ, kb), acc0);
    acc1 = wmma32(pf, load_frag_b(vtb + 16 * SEQ, SEQ, kb), acc1);
    acc2 = wmma32(pf, load_frag_b(vtb + 32 * SEQ, SEQ, kb), acc2);
    acc3 = wmma32(pf, load_frag_b(vtb + 48 * SEQ, SEQ, kb), acc3);
  }

  // normalize + store ctx (f16) in token-major layout for final projection
#pragma unroll
  for (int v = 0; v < 8; ++v) {
    float rl = 1.f / lrun[v];
    int row = q0 + v + 8 * hi;
    _Float16* o = ctx + ((size_t)b * SEQ + row) * DM + h * DKH + ln;
    o[0]  = (_Float16)(acc0[v] * rl);
    o[16] = (_Float16)(acc1[v] * rl);
    o[32] = (_Float16)(acc2[v] * rl);
    o[48] = (_Float16)(acc3[v] * rl);
  }
}

extern "C" void kernel_launch(void* const* d_in, const int* in_sizes, int n_in,
                              void* d_out, int out_size, void* d_ws, size_t ws_size,
                              hipStream_t stream) {
  const float* q   = (const float*)d_in[0];
  const float* k   = (const float*)d_in[1];
  const float* v   = (const float*)d_in[2];
  const int*  mask = (const int*)d_in[3];
  const float* Wq = (const float*)d_in[4],  *Aq = (const float*)d_in[5],  *Bq = (const float*)d_in[6];
  const float* Wk = (const float*)d_in[7],  *Ak = (const float*)d_in[8],  *Bk = (const float*)d_in[9];
  const float* Wv = (const float*)d_in[10], *Av = (const float*)d_in[11], *Bv = (const float*)d_in[12];
  const float* Wo = (const float*)d_in[13], *Ao = (const float*)d_in[14], *Bo = (const float*)d_in[15];

  char* ws = (char*)d_ws;
  const size_t MB = 1024 * 1024;
  _Float16* WEq = (_Float16*)(ws + 0 * MB);
  _Float16* WEk = (_Float16*)(ws + 2 * MB);
  _Float16* WEv = (_Float16*)(ws + 4 * MB);
  _Float16* WEo = (_Float16*)(ws + 6 * MB);
  _Float16* XB  = (_Float16*)(ws + 8 * MB);   // reused for q/k/v inputs
  _Float16* QF  = (_Float16*)(ws + 16 * MB);
  _Float16* KF  = (_Float16*)(ws + 24 * MB);
  _Float16* VTM = (_Float16*)(ws + 32 * MB);  // V projection (pre-transpose)
  _Float16* VT  = (_Float16*)(ws + 40 * MB);  // V transposed [b*DM+d][s]
  _Float16* CTX = (_Float16*)(ws + 32 * MB);  // aliases VTM (dead after transpose)
  unsigned* MTB = (unsigned*)(ws + 48 * MB);  // packed transposed mask (1MB)

  // 0) pack mask bits (transposed for attention access pattern)
  pack_mask<<<dim3(SEQ / 256, QW, NB), 256, 0, stream>>>(mask, MTB);

  // 1) fold LoRA into weights
  fuse_w<<<4096, 256, 0, stream>>>(Wq, Aq, Bq, WEq);
  fuse_w<<<4096, 256, 0, stream>>>(Wk, Ak, Bk, WEk);
  fuse_w<<<4096, 256, 0, stream>>>(Wv, Av, Bv, WEv);
  fuse_w<<<4096, 256, 0, stream>>>(Wo, Ao, Bo, WEo);

  dim3 ggrid(DM / 128, MTOK / 128);
  // 2) projections
  cvt_f16<<<4096, 256, 0, stream>>>(q, XB);
  gemm_nt<0><<<ggrid, 256, 0, stream>>>(XB, WEq, nullptr, QF, MTOK, DM, DM);
  cvt_f16<<<4096, 256, 0, stream>>>(k, XB);
  gemm_nt<0><<<ggrid, 256, 0, stream>>>(XB, WEk, nullptr, KF, MTOK, DM, DM);
  cvt_f16<<<4096, 256, 0, stream>>>(v, XB);
  gemm_nt<0><<<ggrid, 256, 0, stream>>>(XB, WEv, nullptr, VTM, MTOK, DM, DM);
  transpose_v<<<dim3(SEQ / 64, DM / 64, NB), 256, 0, stream>>>(VTM, VT);

  // 3) flash attention
  attn<<<dim3(SEQ / 128, NH, NB), 256, 0, stream>>>(QF, KF, VT, MTB, CTX);

  // 4) output projection -> f32
  gemm_nt<1><<<ggrid, 256, 0, stream>>>(CTX, WEo, (float*)d_out, nullptr, MTOK, DM, DM);
}
